// MultiHeadAttentionLayer_67422396613262
// MI455X (gfx1250) — compile-verified, matrix operand order checked by reading, not run on hardware
//
#include <hip/hip_runtime.h>

// ---------------------------------------------------------------------------
// MultiHeadAttention for MI455X (gfx1250), bf16 WMMA w/ f32 accumulation.
// B=2, L=4096, C=512, H=8, D=64. Compute-bound (~86 GFLOP vs ~34MB HBM I/O).
// - GEMMs: 8-wave blocks; per-K-chunk B tile (128x32) staged once per block
//   into LDS via double-buffered GLOBAL_LOAD_ASYNC_TO_LDS_B128 (ASYNCcnt);
//   A fragments pipelined from global; B fragments depth-2 pipelined from LDS
//   so WMMAs see partial s_wait_dscnt instead of full drains.
// - Attention: flash-attention with async double-buffered K/V LDS tiles.
// ---------------------------------------------------------------------------

#define DEVI __device__ __forceinline__

typedef __bf16 bf16_t;
typedef __attribute__((ext_vector_type(8)))  __bf16 v8bf;
typedef __attribute__((ext_vector_type(16))) __bf16 v16bf;
typedef __attribute__((ext_vector_type(8)))  float  v8f;
typedef __attribute__((ext_vector_type(4)))  int    v4i;

#if defined(__has_builtin)
#if __has_builtin(__builtin_amdgcn_global_load_async_to_lds_b128) && \
    __has_builtin(__builtin_amdgcn_s_wait_asynccnt)
#define USE_ASYNC_LDS 1
#endif
#endif
#ifndef USE_ASYNC_LDS
#define USE_ASYNC_LDS 0
#endif

constexpr int B_  = 2;
constexpr int L_  = 4096;
constexpr int C_  = 512;
constexpr int H_  = 8;
constexpr int D_  = 64;
constexpr int C3_ = 3 * C_;
constexpr float SCALE_ = 22.627416997969522f; // sqrt(512): nonstandard scale from ref

DEVI v8f zero8() {
  v8f z;
#pragma unroll
  for (int i = 0; i < 8; ++i) z[i] = 0.f;
  return z;
}

DEVI v8f wmma_bf16(v16bf a, v16bf b, v8f c) {
  // D = A(16x32 bf16) * B(32x16 bf16) + C(16x16 f32)
  return __builtin_amdgcn_wmma_f32_16x16x32_bf16(
      /*neg_a=*/false, a, /*neg_b=*/false, b,
      /*c_mod=*/(short)0, c, /*reuse_a=*/false, /*reuse_b=*/false);
}

// Load a 16x32 A fragment (or symmetric 32x16 B fragment; "row" index is the
// output column / storage row of the pre-transposed B).
// ISA layout: lanes 0-15 hold row (lane&15) K={0..7,16..23};
//             lanes 16-31 hold the same rows, K={8..15,24..31}.
DEVI v16bf load_frag(const bf16_t* base, int ld) {
  const int lane = threadIdx.x & 31;
  const bf16_t* p = base + (size_t)(lane & 15) * ld + ((lane >> 4) << 3);
  v8bf lo = *reinterpret_cast<const v8bf*>(p);
  v8bf hi = *reinterpret_cast<const v8bf*>(p + 16);
  v16bf f;
#pragma unroll
  for (int i = 0; i < 8; ++i) { f[i] = lo[i]; f[i + 8] = hi[i]; }
  return f;
}

// Row-wise reductions over the C/D layout (row lives across a 16-lane half).
DEVI float redmax16(float v) {
#pragma unroll
  for (int m = 8; m >= 1; m >>= 1) v = fmaxf(v, __shfl_xor(v, m, 32));
  return v;
}
DEVI float redsum16(float v) {
#pragma unroll
  for (int m = 8; m >= 1; m >>= 1) v += __shfl_xor(v, m, 32);
  return v;
}

#if USE_ASYNC_LDS
typedef __attribute__((address_space(1))) v4i* gv4i_p;  // v4i __device__*
typedef __attribute__((address_space(3))) v4i* lv4i_p;  // v4i __shared__*
DEVI void async_b128(const bf16_t* g, bf16_t* l) {
  __builtin_amdgcn_global_load_async_to_lds_b128((gv4i_p)(g), (lv4i_p)(l), 0, 0);
}
#endif

// ---------------------------------------------------------------------------
// Kernel 1: fp32 -> bf16 packing; weights transposed to [N][K] so B fragments
// are contiguous b128 loads per lane.
// ---------------------------------------------------------------------------
__global__ void pack_kernel(const float* __restrict__ x,
                            const float* __restrict__ wkqv,
                            const float* __restrict__ wout,
                            bf16_t* __restrict__ xbf,
                            bf16_t* __restrict__ wkqvT,
                            bf16_t* __restrict__ woutT) {
  size_t i = (size_t)blockIdx.x * blockDim.x + threadIdx.x;
  const size_t nx = (size_t)B_ * L_ * C_;
  if (i < nx) xbf[i] = (bf16_t)x[i];
  if (i < (size_t)C_ * C3_) {
    int k = (int)(i / C3_), n = (int)(i % C3_);
    wkqvT[(size_t)n * C_ + k] = (bf16_t)wkqv[i];
  }
  if (i < (size_t)C_ * C_) {
    int k = (int)(i / C_), n = (int)(i % C_);
    woutT[(size_t)n * C_ + k] = (bf16_t)wout[i];
  }
}

// ---------------------------------------------------------------------------
// Shared GEMM core: 8 waves x (16x128) tiles over one 128-col B block.
// B K-chunk tile [128 cols][32 k] (8KB) double-buffered in LDS, staged once
// per workgroup per chunk; A fragments pipelined from global per wave;
// B fragments depth-2 pipelined from LDS.
// ---------------------------------------------------------------------------
DEVI void gemm_block_16x128(const bf16_t* __restrict__ Abase,
                            const bf16_t* __restrict__ Bbase,  // wT + col0*C_
                            bf16_t* __restrict__ sB,           // [2][128*32]
                            v8f acc[8]) {
  const int tid = threadIdx.x;
  const int scol = tid >> 1, sseg = tid & 1;  // 256 threads: 2 per B column
  constexpr int NC = C_ / 32;                 // 16 K-chunks

#if USE_ASYNC_LDS
  auto stageB = [&](int kc, int buf) {
    const bf16_t* src = Bbase + (size_t)scol * C_ + kc + sseg * 16;
    bf16_t* dst = sB + buf * (128 * 32) + scol * 32 + sseg * 16;
    async_b128(src, dst);
    async_b128(src + 8, dst + 8);
  };
  stageB(0, 0);
#endif
  v16bf af = load_frag(Abase, C_);

  for (int t = 0; t < NC; ++t) {
    const int buf = t & 1;
#if USE_ASYNC_LDS
    if (t + 1 < NC) {
      // Buffer buf^1 was last read in iteration t-1; the end-of-iteration
      // barrier guarantees all waves are done with it.
      stageB(32 * (t + 1), buf ^ 1);
      __builtin_amdgcn_s_wait_asynccnt(2);  // this tile's 2 ops done (in order)
    } else {
      __builtin_amdgcn_s_wait_asynccnt(0);
    }
    __syncthreads();
#else
    {
      const v8bf* src = reinterpret_cast<const v8bf*>(
          Bbase + (size_t)scol * C_ + 32 * t + sseg * 16);
      v8bf a0 = src[0], a1 = src[1];
      v8bf* dst = reinterpret_cast<v8bf*>(sB + buf * (128 * 32) + scol * 32 +
                                          sseg * 16);
      dst[0] = a0; dst[1] = a1;
    }
    __syncthreads();
#endif
    const bf16_t* sBbuf = sB + buf * (128 * 32);

    // Prefetch next A fragment from global first: its latency is hidden by
    // the whole WMMA batch below.
    v16bf afn = af;
    if (t + 1 < NC) afn = load_frag(Abase + 32 * (t + 1), C_);

    // Depth-2 rotating pipeline of B fragments from LDS: each ds_load has
    // two WMMAs of slack before its data is consumed.
    v16bf bA = load_frag(sBbuf + 0 * (16 * 32), 32);
    v16bf bB = load_frag(sBbuf + 1 * (16 * 32), 32);
#pragma unroll
    for (int nb = 0; nb < 8; ++nb) {
      v16bf bC = bA;
      if (nb + 2 < 8) bC = load_frag(sBbuf + (nb + 2) * (16 * 32), 32);
      acc[nb] = wmma_bf16(af, bA, acc[nb]);
      bA = bB;
      bB = bC;
    }
    af = afn;

    __syncthreads();  // all waves done reading this LDS buffer
  }
}

// ---------------------------------------------------------------------------
// Kernel 2: kqv = x @ w_kqv. Epilogue scatters bf16 into k [BH,L,D],
// q [BH,L,D], v-transposed [BH,D,L].
// ---------------------------------------------------------------------------
__global__ __launch_bounds__(256, 1) void gemm_kqv_kernel(
    const bf16_t* __restrict__ xbf, const bf16_t* __restrict__ wT,
    bf16_t* __restrict__ kbuf, bf16_t* __restrict__ qbuf,
    bf16_t* __restrict__ vbuf) {
  __shared__ bf16_t sB[2][128 * 32];
  const int lane = threadIdx.x & 31, wave = threadIdx.x >> 5;
  const int row0 = (blockIdx.x * 8 + wave) * 16;
  const int col0 = blockIdx.y * 128;
  v8f acc[8];
#pragma unroll
  for (int nb = 0; nb < 8; ++nb) acc[nb] = zero8();

  gemm_block_16x128(xbf + (size_t)row0 * C_, wT + (size_t)col0 * C_, &sB[0][0],
                    acc);

  const int half = lane >> 4, cl = lane & 15;
#pragma unroll
  for (int nb = 0; nb < 8; ++nb) {
    const int colbase = col0 + 16 * nb;
    const int reg = colbase >> 9;  // 0=k, 1=q, 2=v (chunk order k,q,v)
    const int h = (colbase >> 6) & 7;
    const int d = (colbase & 63) + cl;
#pragma unroll
    for (int j = 0; j < 8; ++j) {
      const int token = row0 + j + 8 * half;
      const int b = token >> 12, l = token & (L_ - 1);
      const size_t bh = (size_t)(b * H_ + h);
      const bf16_t v = (bf16_t)acc[nb][j];
      if (reg == 0)      kbuf[(bh * L_ + l) * D_ + d] = v;
      else if (reg == 1) qbuf[(bh * L_ + l) * D_ + d] = v;
      else               vbuf[(bh * D_ + d) * L_ + l] = v;  // V transposed
    }
  }
}

// ---------------------------------------------------------------------------
// Kernel 3: flash attention per (b,h). Workgroup = 4 waves x 16 query rows.
// K[32][64] + Vt[64][32] tiles double-buffered in LDS via async-to-LDS.
// ---------------------------------------------------------------------------
__global__ __launch_bounds__(128, 1) void attn_kernel(
    const bf16_t* __restrict__ qbuf, const bf16_t* __restrict__ kbuf,
    const bf16_t* __restrict__ vbuf, bf16_t* __restrict__ attnbuf) {
  __shared__ bf16_t sK[2][32 * 64];   // [buf][key][d]
  __shared__ bf16_t sV[2][64 * 32];   // [buf][d][key] (V transposed)
  __shared__ bf16_t sP[4][16 * 32];   // per-wave P relayout scratch

  const int tid = threadIdx.x;
  const int lane = tid & 31, w = tid >> 5;
  const int half = lane >> 4, cl = lane & 15;
  const int bh = blockIdx.y;
  const int q0 = blockIdx.x * 64 + w * 16;

  const bf16_t* qg = qbuf + ((size_t)bh * L_ + q0) * D_;
  const v16bf Q0 = load_frag(qg + 0, D_);   // d 0..31
  const v16bf Q1 = load_frag(qg + 32, D_);  // d 32..63

  float mrun[8], lrun[8];
  v8f acc[4];
#pragma unroll
  for (int j = 0; j < 8; ++j) { mrun[j] = -1e30f; lrun[j] = 0.f; }
#pragma unroll
  for (int nc = 0; nc < 4; ++nc) acc[nc] = zero8();

  const bf16_t* kg_base = kbuf + (size_t)bh * L_ * D_;
  const bf16_t* vg_base = vbuf + (size_t)bh * D_ * L_;

  const int dr = tid >> 1, seg = tid & 1;  // V staging: 2 threads per d-row

  constexpr int NT = L_ / 32;  // 128 key tiles

#if USE_ASYNC_LDS
  // 4 async b128 per thread per tile: 2 for K (16 halfs), 2 for Vt (16 halfs).
  auto stage_async = [&](int key0, int buf) {
    const bf16_t* ksrc = kg_base + (size_t)key0 * D_ + 16 * tid;
    bf16_t* kdst = &sK[buf][16 * tid];
    async_b128(ksrc, kdst);
    async_b128(ksrc + 8, kdst + 8);
    const bf16_t* vsrc = vg_base + (size_t)dr * L_ + key0 + seg * 16;
    bf16_t* vdst = &sV[buf][dr * 32 + seg * 16];
    async_b128(vsrc, vdst);
    async_b128(vsrc + 8, vdst + 8);
  };
  stage_async(0, 0);
#endif

  for (int t = 0; t < NT; ++t) {
    const int key0 = 32 * t;
    const int buf = t & 1;

#if USE_ASYNC_LDS
    if (t + 1 < NT) {
      stage_async(32 * (t + 1), (t + 1) & 1);
      __builtin_amdgcn_s_wait_asynccnt(4);  // tile t's 4 ops done (in order)
    } else {
      __builtin_amdgcn_s_wait_asynccnt(0);
    }
    __syncthreads();
#else
    {
      const v8bf* src =
          reinterpret_cast<const v8bf*>(kg_base + (size_t)key0 * D_);
      v8bf* dst = reinterpret_cast<v8bf*>(sK[buf]);
      v8bf t0 = src[2 * tid], t1 = src[2 * tid + 1];
      dst[2 * tid] = t0; dst[2 * tid + 1] = t1;
      const v8bf* vs = reinterpret_cast<const v8bf*>(
          vg_base + (size_t)dr * L_ + key0 + seg * 16);
      v8bf u0 = vs[0], u1 = vs[1];
      v8bf* vd = reinterpret_cast<v8bf*>(&sV[buf][dr * 32 + seg * 16]);
      vd[0] = u0; vd[1] = u1;
    }
    __syncthreads();
#endif

    // --- batch all LDS fragment loads (distinct regs -> pipelined ds) ------
    v16bf kb[2][2], vb[4];
#pragma unroll
    for (int s = 0; s < 2; ++s)
#pragma unroll
      for (int c = 0; c < 2; ++c)
        kb[s][c] = load_frag(&sK[buf][16 * s * 64 + 32 * c], 64);
#pragma unroll
    for (int nc = 0; nc < 4; ++nc)
      vb[nc] = load_frag(&sV[buf][16 * nc * 32], 32);

    // --- scores: S[16 q rows][32 keys] = Q(16x64) * K^T --------------------
    v8f S[2];
#pragma unroll
    for (int s = 0; s < 2; ++s) {
      v8f z = zero8();
      z = wmma_bf16(Q0, kb[s][0], z);
      z = wmma_bf16(Q1, kb[s][1], z);
#pragma unroll
      for (int j = 0; j < 8; ++j) z[j] *= SCALE_;
      S[s] = z;
    }

    // --- online softmax (C layout: slot j -> row j+8*half, lane%16 = col) --
#pragma unroll
    for (int j = 0; j < 8; ++j) {
      float cm = redmax16(fmaxf(S[0][j], S[1][j]));
      float mnew = fmaxf(mrun[j], cm);
      float alpha = __expf(mrun[j] - mnew);
      float p0 = __expf(S[0][j] - mnew);
      float p1 = __expf(S[1][j] - mnew);
      float rs = redsum16(p0 + p1);
      lrun[j] = lrun[j] * alpha + rs;
      mrun[j] = mnew;
#pragma unroll
      for (int nc = 0; nc < 4; ++nc) acc[nc][j] *= alpha;
      const int row = j + 8 * half;
      sP[w][row * 32 + cl]      = (bf16_t)p0;
      sP[w][row * 32 + cl + 16] = (bf16_t)p1;
    }

    // --- P (A-frag via per-wave LDS bounce; same-wave DS is in order) ------
    v16bf pf = load_frag(sP[w], 32);
#pragma unroll
    for (int nc = 0; nc < 4; ++nc)
      acc[nc] = wmma_bf16(pf, vb[nc], acc[nc]);

    __syncthreads();  // everyone done reading this tile's LDS buffers
  }

  // --- normalize and emit bf16 [B*L, C] for the output projection ----------
  const int b = bh >> 3, h = bh & 7;
#pragma unroll
  for (int j = 0; j < 8; ++j) {
    const float inv = 1.f / lrun[j];
    const int token = b * L_ + q0 + j + 8 * half;
#pragma unroll
    for (int nc = 0; nc < 4; ++nc) {
      const int col = h * 64 + 16 * nc + cl;
      attnbuf[(size_t)token * C_ + col] = (bf16_t)(acc[nc][j] * inv);
    }
  }
}

// ---------------------------------------------------------------------------
// Kernel 4: out = attn @ w_out + b_out, fp32 output.
// ---------------------------------------------------------------------------
__global__ __launch_bounds__(256, 1) void gemm_out_kernel(
    const bf16_t* __restrict__ attnbf, const bf16_t* __restrict__ woutT,
    const float* __restrict__ bias, float* __restrict__ out) {
  __shared__ bf16_t sB[2][128 * 32];
  const int lane = threadIdx.x & 31, wave = threadIdx.x >> 5;
  const int row0 = (blockIdx.x * 8 + wave) * 16;
  const int col0 = blockIdx.y * 128;
  v8f acc[8];
#pragma unroll
  for (int nb = 0; nb < 8; ++nb) acc[nb] = zero8();

  gemm_block_16x128(attnbf + (size_t)row0 * C_, woutT + (size_t)col0 * C_,
                    &sB[0][0], acc);

  const int half = lane >> 4, cl = lane & 15;
#pragma unroll
  for (int nb = 0; nb < 8; ++nb) {
    const int g = col0 + 16 * nb + cl;
    const float bv = bias[g];
#pragma unroll
    for (int j = 0; j < 8; ++j) {
      const int token = row0 + j + 8 * half;
      out[(size_t)token * C_ + g] = acc[nb][j] + bv;
    }
  }
}

// ---------------------------------------------------------------------------
extern "C" void kernel_launch(void* const* d_in, const int* in_sizes, int n_in,
                              void* d_out, int out_size, void* d_ws, size_t ws_size,
                              hipStream_t stream) {
  const float* x     = (const float*)d_in[0];
  const float* w_kqv = (const float*)d_in[1];
  const float* w_out = (const float*)d_in[2];
  const float* b_out = (const float*)d_in[3];
  float* out = (float*)d_out;
  (void)in_sizes; (void)n_in; (void)out_size; (void)ws_size;

  char* ws = (char*)d_ws;
  size_t off = 0;
  auto walloc = [&](size_t bytes) -> void* {
    void* p = ws + off;
    off += (bytes + 255) & ~size_t(255);
    return p;
  };
  // ~42 MB of scratch total
  bf16_t* xbf   = (bf16_t*)walloc((size_t)B_ * L_ * C_ * 2);
  bf16_t* wkqvT = (bf16_t*)walloc((size_t)C3_ * C_ * 2);
  bf16_t* woutT = (bf16_t*)walloc((size_t)C_ * C_ * 2);
  bf16_t* kbuf  = (bf16_t*)walloc((size_t)B_ * H_ * L_ * D_ * 2);
  bf16_t* qbuf  = (bf16_t*)walloc((size_t)B_ * H_ * L_ * D_ * 2);
  bf16_t* vbuf  = (bf16_t*)walloc((size_t)B_ * H_ * L_ * D_ * 2);
  bf16_t* attnb = (bf16_t*)walloc((size_t)B_ * L_ * C_ * 2);

  const int npack = B_ * L_ * C_;
  pack_kernel<<<(npack + 255) / 256, 256, 0, stream>>>(x, w_kqv, w_out, xbf,
                                                       wkqvT, woutT);

  // 512 M-tiles of 16 rows, 8 waves per block sharing one 128-col B tile
  gemm_kqv_kernel<<<dim3((B_ * L_ / 16) / 8, C3_ / 128), 256, 0, stream>>>(
      xbf, wkqvT, kbuf, qbuf, vbuf);

  attn_kernel<<<dim3(L_ / 64, B_ * H_), 128, 0, stream>>>(qbuf, kbuf, vbuf,
                                                          attnb);

  gemm_out_kernel<<<dim3((B_ * L_ / 16) / 8, C_ / 128), 256, 0, stream>>>(
      attnb, woutT, b_out, out);
}
